// RelationAttention_76922864271958
// MI455X (gfx1250) — compile-verified, matrix operand order checked by reading, not run on hardware
//
#include <hip/hip_runtime.h>
#include <hip/hip_bf16.h>

// ---------------------------------------------------------------------------
// RelationAttention on MI455X (gfx1250), wave32, bf16 WMMA with f32 accum.
//
// N=M=2048 tokens, APP_IN=512, H=8 heads, dh=128, dv=64.  ~31 GFLOP over
// ~30MB (AI ~1000 FLOP/B) -> compute bound; everything routed through
// v_wmma_f32_16x16x32_bf16, all operand fetches as contiguous 16B b128 loads.
//
// Stages:
//   1) cvt_x:  activations f32 -> bf16 (vectorized, once).
//   2) wt:     weights f32 [512,C] -> bf16 W^T [C,512] via LDS 32x32 tile
//              (coalesced both sides) so GEMM B-fragments are contiguous.
//   3) proj:   X @ W + b, wave = 16 rows x 64 cols (4 WMMA tiles share one
//              A-fragment), 64 WMMAs over K=512; scatter head-major (K,G1,Q,G2)
//              or transposed (V^T).
//   4) norm:   per-row L2 normalize Q, G2 (wave/row, shuffle reduce).
//   5) attn:   transposed-score trick: S^T = K*Q^T, G^T = G1*G2^T,
//              s = exp(S^T*G^T/dh^2), O^T += V^T*s.  The f32 D layout
//              (lane = query col, VGPR r = key r+8*half) packs lane-locally
//              into the next WMMA's bf16 B-fragment: no LDS, no permutes.
// ---------------------------------------------------------------------------

typedef __attribute__((ext_vector_type(16))) __bf16 v16bf;
typedef __attribute__((ext_vector_type(8)))  __bf16 v8bf;
typedef __attribute__((ext_vector_type(8)))  float  v8f;

#define HEADS 8
#define NTOK  2048
#define MTOK  2048
#define DIN   512
#define DH    128
#define DV    64

// 16-element bf16 fragment = two contiguous 16-byte chunks (CDNA5 16-bit A/B
// layout: per lane, K values half*8+[0..7] in elems 0..7, 16+half*8+[0..7]
// in elems 8..15).
__device__ __forceinline__ v16bf load_frag(const __bf16* p0, const __bf16* p1) {
  v8bf lo = *(const v8bf*)p0;
  v8bf hi = *(const v8bf*)p1;
  v16bf r;
#pragma unroll
  for (int i = 0; i < 8; ++i) { r[i] = lo[i]; r[8 + i] = hi[i]; }
  return r;
}

__device__ __forceinline__ v8f wmma_bf16(v16bf a, v16bf b, v8f c) {
  return __builtin_amdgcn_wmma_f32_16x16x32_bf16(
      false, a, false, b, (short)0, c, false, false);
}

// ---------------------------------------------------------------------------
// Activations f32 -> bf16, 8 elems/thread (b256 f32 in, b128 bf16 out).
// ---------------------------------------------------------------------------
__global__ __launch_bounds__(256) void cvt_x_kernel(const float* __restrict__ X,
                                                    __bf16* __restrict__ out) {
  const size_t i = ((size_t)blockIdx.x * 256 + threadIdx.x) * 8;
  v8f x = *(const v8f*)(X + i);
  v8bf o;
#pragma unroll
  for (int j = 0; j < 8; ++j) o[j] = (__bf16)x[j];
  *(v8bf*)(out + i) = o;
}

// ---------------------------------------------------------------------------
// Weight transpose+convert: W[512,C] f32 -> Wt[C,512] bf16.  32x32 LDS tile,
// coalesced global read and write.
// ---------------------------------------------------------------------------
__global__ __launch_bounds__(256) void wt_kernel(const float* __restrict__ W,
                                                 __bf16* __restrict__ Wt, int C) {
  __shared__ float tile[32][33];
  const int t = threadIdx.x;
  const int kbase = blockIdx.y * 32;   // row in W (feature, 0..511)
  const int cbase = blockIdx.x * 32;   // col in W (output channel)
  const int c = t & 31, r0 = t >> 5;
#pragma unroll
  for (int i = 0; i < 4; ++i)
    tile[r0 + i * 8][c] = W[(size_t)(kbase + r0 + i * 8) * C + cbase + c];
  __syncthreads();
  const int kk = t & 31, cc0 = t >> 5;
#pragma unroll
  for (int i = 0; i < 4; ++i)
    Wt[(size_t)(cbase + cc0 + i * 8) * DIN + kbase + kk] =
        (__bf16)tile[kk][cc0 + i * 8];
}

// ---------------------------------------------------------------------------
// Projection GEMM: out = X[2048,512] @ W[512,C] + b (both bf16, f32 accum).
// Wave = 16 rows x 64 cols: one A-fragment feeds 4 WMMAs per K-chunk.
// mode 0: head-major [h][row][c%128] (C=1024);  mode 1: V^T [h][c%64][row].
// ---------------------------------------------------------------------------
__global__ __launch_bounds__(256) void proj_kernel(
    const __bf16* __restrict__ Xb, const __bf16* __restrict__ Wt,
    const float* __restrict__ b, __bf16* __restrict__ out,
    int C, int mode) {
  const int lane = threadIdx.x & 31;
  const int wave = threadIdx.x >> 5;
  const int half = lane >> 4;
  const int idx  = lane & 15;
  const int rbase = blockIdx.x * 16;
  const int cbase = (blockIdx.y * 8 + wave) * 64;

  v8f acc0 = {}, acc1 = {}, acc2 = {}, acc3 = {};
#pragma unroll 2
  for (int kc = 0; kc < DIN; kc += 32) {
    const __bf16* px = Xb + (size_t)(rbase + idx) * DIN + kc + half * 8;
    v16bf af = load_frag(px, px + 16);
    const __bf16* pw = Wt + (size_t)(cbase + idx) * DIN + kc + half * 8;
    acc0 = wmma_bf16(af, load_frag(pw, pw + 16), acc0);
    pw += (size_t)16 * DIN;
    acc1 = wmma_bf16(af, load_frag(pw, pw + 16), acc1);
    pw += (size_t)16 * DIN;
    acc2 = wmma_bf16(af, load_frag(pw, pw + 16), acc2);
    pw += (size_t)16 * DIN;
    acc3 = wmma_bf16(af, load_frag(pw, pw + 16), acc3);
  }
#pragma unroll
  for (int t = 0; t < 4; ++t) {
    const v8f acc = (t == 0) ? acc0 : (t == 1) ? acc1 : (t == 2) ? acc2 : acc3;
    const int col = cbase + t * 16 + idx;
    const float bias = b[col];
#pragma unroll
    for (int r = 0; r < 8; ++r) {
      const int row = rbase + half * 8 + r;   // D layout: M = r + 8*half
      const float v = acc[r] + bias;
      if (mode == 0) {
        const int h = col >> 7, off = col & 127;
        out[((size_t)h * NTOK + row) * DH + off] = (__bf16)v;
      } else {
        const int h = col >> 6, vd = col & 63;
        out[(size_t)(h * DV + vd) * NTOK + row] = (__bf16)v;   // V^T
      }
    }
  }
}

// ---------------------------------------------------------------------------
// Row L2-normalization of Q and G2 ([8][2048][128] bf16, in place).
// ---------------------------------------------------------------------------
__global__ __launch_bounds__(256) void norm_kernel(__bf16* __restrict__ Qb,
                                                   __bf16* __restrict__ G2b) {
  const int lane  = threadIdx.x & 31;
  const int wave  = threadIdx.x >> 5;
  const int rowid = blockIdx.x * 8 + wave;
  __bf16* base = (rowid < HEADS * NTOK)
                     ? Qb + (size_t)rowid * DH
                     : G2b + (size_t)(rowid - HEADS * NTOK) * DH;
  float v[4];
  float ss = 0.0f;
#pragma unroll
  for (int i = 0; i < 4; ++i) {
    v[i] = (float)base[lane * 4 + i];
    ss += v[i] * v[i];
  }
#pragma unroll
  for (int m = 1; m <= 16; m <<= 1) ss += __shfl_xor(ss, m, 32);
  const float inv = rsqrtf(ss);
#pragma unroll
  for (int i = 0; i < 4; ++i) base[lane * 4 + i] = (__bf16)(v[i] * inv);
}

// ---------------------------------------------------------------------------
// Attention: per wave, 16 queries of one head; stream keys in chunks of 32.
// 20 WMMAs per chunk (8 S^T, 8 G^T, 4 O^T), all loads contiguous b128.
// ---------------------------------------------------------------------------
__global__ __launch_bounds__(256) void attn_kernel(
    const __bf16* __restrict__ Kb, const __bf16* __restrict__ G1b,
    const __bf16* __restrict__ Qb, const __bf16* __restrict__ G2b,
    const __bf16* __restrict__ Vt, float* __restrict__ out) {
  const int lane = threadIdx.x & 31;
  const int wave = threadIdx.x >> 5;
  const int half = lane >> 4;
  const int idx  = lane & 15;
  const int h     = blockIdx.y;
  const int qbase = blockIdx.x * 128 + wave * 16;

  const __bf16* Qh  = Qb  + (size_t)h * NTOK * DH;
  const __bf16* G2h = G2b + (size_t)h * NTOK * DH;
  const __bf16* Kh  = Kb  + (size_t)h * NTOK * DH;
  const __bf16* G1h = G1b + (size_t)h * NTOK * DH;
  const __bf16* Vh  = Vt  + (size_t)h * DV * NTOK;

  // Resident B-fragments for Q^T and G2^T (lane idx = query column).
  v16bf qf[4], gf[4];
#pragma unroll
  for (int kc = 0; kc < 4; ++kc) {
    const __bf16* pq = Qh + (size_t)(qbase + idx) * DH + kc * 32 + half * 8;
    qf[kc] = load_frag(pq, pq + 16);
    const __bf16* pg = G2h + (size_t)(qbase + idx) * DH + kc * 32 + half * 8;
    gf[kc] = load_frag(pg, pg + 16);
  }

  v8f o0 = {}, o1 = {}, o2 = {}, o3 = {};
  float den = 0.0f;
  const float SCALE = 1.0f / (128.0f * 128.0f);  // exp((QK/dh)*(G2G1/dh))

  for (int nb = 0; nb < NTOK; nb += 32) {
    v8f st0 = {}, st1 = {}, gt0 = {}, gt1 = {};
#pragma unroll
    for (int kc = 0; kc < 4; ++kc) {
      const int fo = kc * 32 + half * 8;
      const __bf16* pk0 = Kh  + (size_t)(nb + idx) * DH + fo;
      const __bf16* pg0 = G1h + (size_t)(nb + idx) * DH + fo;
      const __bf16* pk1 = Kh  + (size_t)(nb + 16 + idx) * DH + fo;
      const __bf16* pg1 = G1h + (size_t)(nb + 16 + idx) * DH + fo;
      st0 = wmma_bf16(load_frag(pk0, pk0 + 16), qf[kc], st0);
      gt0 = wmma_bf16(load_frag(pg0, pg0 + 16), gf[kc], gt0);
      st1 = wmma_bf16(load_frag(pk1, pk1 + 16), qf[kc], st1);
      gt1 = wmma_bf16(load_frag(pg1, pg1 + 16), gf[kc], gt1);
    }
    // exp + denominator + lane-local repack of S^T D-tiles into B-fragment:
    // D row r of key-tile t == key nb+16t+8*half+r, exactly B-frag elem t*8+r.
    v16bf sb;
#pragma unroll
    for (int r = 0; r < 8; ++r) {
      const float s0 = __expf(st0[r] * gt0[r] * SCALE);
      const float s1 = __expf(st1[r] * gt1[r] * SCALE);
      den += s0 + s1;
      sb[r]     = (__bf16)s0;
      sb[8 + r] = (__bf16)s1;
    }
    const __bf16* pv0 = Vh + (size_t)(0 * 16 + idx) * NTOK + nb + half * 8;
    const __bf16* pv1 = Vh + (size_t)(1 * 16 + idx) * NTOK + nb + half * 8;
    const __bf16* pv2 = Vh + (size_t)(2 * 16 + idx) * NTOK + nb + half * 8;
    const __bf16* pv3 = Vh + (size_t)(3 * 16 + idx) * NTOK + nb + half * 8;
    o0 = wmma_bf16(load_frag(pv0, pv0 + 16), sb, o0);
    o1 = wmma_bf16(load_frag(pv1, pv1 + 16), sb, o1);
    o2 = wmma_bf16(load_frag(pv2, pv2 + 16), sb, o2);
    o3 = wmma_bf16(load_frag(pv3, pv3 + 16), sb, o3);
  }

  den += __shfl_xor(den, 16, 32);       // combine key-halves per query
  const float inv = 1.0f / den;
  const int q = qbase + idx;            // D layout: lane idx = query column
  float* orow = out + (size_t)q * (HEADS * DV) + h * DV + half * 8;
#pragma unroll
  for (int r = 0; r < 8; ++r) {         // vd = vt*16 + 8*half + r: contiguous
    orow[0 * 16 + r] = o0[r] * inv;
    orow[1 * 16 + r] = o1[r] * inv;
    orow[2 * 16 + r] = o2[r] * inv;
    orow[3 * 16 + r] = o3[r] * inv;
  }
}

// ---------------------------------------------------------------------------
extern "C" void kernel_launch(void* const* d_in, const int* in_sizes, int n_in,
                              void* d_out, int out_size, void* d_ws, size_t ws_size,
                              hipStream_t stream) {
  const float* first_app  = (const float*)d_in[0];
  const float* first_pos  = (const float*)d_in[1];
  const float* second_app = (const float*)d_in[2];
  const float* second_pos = (const float*)d_in[3];
  const float* WK  = (const float*)d_in[4];   const float* bK  = (const float*)d_in[5];
  const float* WQ  = (const float*)d_in[6];   const float* bQ  = (const float*)d_in[7];
  const float* WV  = (const float*)d_in[8];   const float* bV  = (const float*)d_in[9];
  const float* WG1 = (const float*)d_in[10];  const float* bG1 = (const float*)d_in[11];
  const float* WG2 = (const float*)d_in[12];  const float* bG2 = (const float*)d_in[13];
  float* out = (float*)d_out;

  // Scratch layout (bf16):
  //   4 x 4MiB head-major projections, 2MiB V^T,
  //   4 x 2MiB bf16 activations, 4 x 1MiB + 0.5MiB bf16 W^T   (~30.5 MiB).
  char* ws = (char*)d_ws;
  const size_t SZ_PROJ = (size_t)HEADS * NTOK * DH * sizeof(__bf16);  // 4 MiB
  const size_t SZ_XBF  = (size_t)NTOK * DIN * sizeof(__bf16);         // 2 MiB
  const size_t SZ_WT   = (size_t)1024 * DIN * sizeof(__bf16);         // 1 MiB
  __bf16* Kb   = (__bf16*)(ws);
  __bf16* G1b  = (__bf16*)(ws + 1 * SZ_PROJ);
  __bf16* Qb   = (__bf16*)(ws + 2 * SZ_PROJ);
  __bf16* G2b  = (__bf16*)(ws + 3 * SZ_PROJ);
  __bf16* Vtb  = (__bf16*)(ws + 4 * SZ_PROJ);                 // 2 MiB
  char* p = ws + 4 * SZ_PROJ + (SZ_PROJ / 2);
  __bf16* Xfa  = (__bf16*)(p);               // first_app  bf16
  __bf16* Xfp  = (__bf16*)(p + 1 * SZ_XBF);  // first_pos  bf16
  __bf16* Xsa  = (__bf16*)(p + 2 * SZ_XBF);  // second_app bf16
  __bf16* Xsp  = (__bf16*)(p + 3 * SZ_XBF);  // second_pos bf16
  char* q = p + 4 * SZ_XBF;
  __bf16* WtK  = (__bf16*)(q);
  __bf16* WtQ  = (__bf16*)(q + 1 * SZ_WT);
  __bf16* WtG1 = (__bf16*)(q + 2 * SZ_WT);
  __bf16* WtG2 = (__bf16*)(q + 3 * SZ_WT);
  __bf16* WtV  = (__bf16*)(q + 4 * SZ_WT);   // 0.5 MiB used

  const dim3 blk(256);
  const int nxb = (NTOK * DIN) / (8 * 256);  // cvt_x grid

  cvt_x_kernel<<<dim3(nxb), blk, 0, stream>>>(first_app,  Xfa);
  cvt_x_kernel<<<dim3(nxb), blk, 0, stream>>>(first_pos,  Xfp);
  cvt_x_kernel<<<dim3(nxb), blk, 0, stream>>>(second_app, Xsa);
  cvt_x_kernel<<<dim3(nxb), blk, 0, stream>>>(second_pos, Xsp);

  wt_kernel<<<dim3(1024 / 32, DIN / 32), blk, 0, stream>>>(WK,  WtK,  1024);
  wt_kernel<<<dim3(1024 / 32, DIN / 32), blk, 0, stream>>>(WQ,  WtQ,  1024);
  wt_kernel<<<dim3(1024 / 32, DIN / 32), blk, 0, stream>>>(WG1, WtG1, 1024);
  wt_kernel<<<dim3(1024 / 32, DIN / 32), blk, 0, stream>>>(WG2, WtG2, 1024);
  wt_kernel<<<dim3( 512 / 32, DIN / 32), blk, 0, stream>>>(WV,  WtV,   512);

  proj_kernel<<<dim3(NTOK / 16, 1024 / 512), blk, 0, stream>>>(Xfa, WtK,  bK,  Kb,  1024, 0);
  proj_kernel<<<dim3(NTOK / 16, 1024 / 512), blk, 0, stream>>>(Xfp, WtG1, bG1, G1b, 1024, 0);
  proj_kernel<<<dim3(MTOK / 16, 1024 / 512), blk, 0, stream>>>(Xsa, WtQ,  bQ,  Qb,  1024, 0);
  proj_kernel<<<dim3(MTOK / 16, 1024 / 512), blk, 0, stream>>>(Xsp, WtG2, bG2, G2b, 1024, 0);
  proj_kernel<<<dim3(NTOK / 16,  512 / 512), blk, 0, stream>>>(Xfa, WtV,  bV,  Vtb,  512, 1);

  norm_kernel<<<dim3((2 * HEADS * NTOK) / 8), blk, 0, stream>>>(Qb, G2b);

  attn_kernel<<<dim3(MTOK / 128, HEADS), blk, 0, stream>>>(Kb, G1b, Qb, G2b, Vtb, out);
}
